// SemanticManifoldOperator_37048387895881
// MI455X (gfx1250) — compile-verified, matrix operand order checked by reading, not run on hardware
//
#include <hip/hip_runtime.h>
#include <hip/hip_bf16.h>
#include <stdint.h>

// Problem constants
constexpr int B_ = 4;
constexpr int N_ = 2048;
constexpr int D_ = 4096;
constexpr int O_ = 4096;
constexpr int R_ = 128;
constexpr int E_ = 4;
constexpr float INV_TEMP = 2.0f; // 1 / 0.5

typedef __attribute__((ext_vector_type(16))) __bf16 v16bf;
typedef __attribute__((ext_vector_type(8)))  __bf16 v8bf;
typedef __attribute__((ext_vector_type(4)))  __bf16 v4bf;
typedef __attribute__((ext_vector_type(8)))  float  v8f;
typedef __attribute__((ext_vector_type(4)))  float  v4f;

__device__ inline v8f v8f_zero() {
    v8f z;
#pragma unroll
    for (int i = 0; i < 8; ++i) z[i] = 0.0f;
    return z;
}

__device__ inline v8f wmma_bf16(v16bf a, v16bf b, v8f c) {
    // D = A(16x32 bf16) x B(32x16 bf16) + C(16x16 f32)
    return __builtin_amdgcn_wmma_f32_16x16x32_bf16(
        /*neg_a=*/false, a, /*neg_b=*/false, b,
        /*c_mod=*/(short)0, c, /*reuse_a=*/false, /*reuse_b=*/false);
}

// CDNA5 async copy: 32 bytes global -> LDS per lane (two B128 transfers).
// INST_OFFSET is added to both the LDS and global addresses (ISA 15.18.3).
__device__ inline void async_load_b128x2(uint32_t lds_off, const void* gptr) {
    asm volatile(
        "global_load_async_to_lds_b128 %0, %1, off\n\t"
        "global_load_async_to_lds_b128 %0, %1, off offset:16"
        :: "v"(lds_off), "v"(gptr) : "memory");
}
__device__ inline void wait_async0() {
    asm volatile("s_wait_asynccnt 0" ::: "memory");
}
__device__ inline uint32_t lds_offset_of(const void* p) {
    // generic LDS pointer: addr[31:0] is the LDS byte address (aperture layout)
    return (uint32_t)(uintptr_t)p;
}

// Load a 16x32 fragment from row-major bf16 memory (K contiguous, 16B-aligned chunks).
// Lane layout (wave32): row = lane&15; k-chunks [kb..kb+7] and [16+kb..16+kb+7], kb=(lane>>4)*8.
__device__ inline v16bf load_frag_bf16(const __bf16* __restrict__ src, int ld) {
    int lane = threadIdx.x & 31;
    const __bf16* p = src + (size_t)(lane & 15) * ld + ((lane >> 4) << 3);
    v8bf lo = *reinterpret_cast<const v8bf*>(p);
    v8bf hi = *reinterpret_cast<const v8bf*>(p + 16);
    v16bf f;
#pragma unroll
    for (int j = 0; j < 8; ++j) { f[j] = lo[j]; f[8 + j] = hi[j]; }
    return f;
}

// -------- Kernel 0: convert input_basis f32 -> bf16 (L2-resident, 1 MB) --------
__global__ __launch_bounds__(256) void convert_basis_kernel(
    const float* __restrict__ src, __bf16* __restrict__ dst) {
    int idx = (blockIdx.x * 256 + threadIdx.x) * 4;
    v4f v = *reinterpret_cast<const v4f*>(src + idx);
    v4bf o;
    o[0] = (__bf16)v[0]; o[1] = (__bf16)v[1]; o[2] = (__bf16)v[2]; o[3] = (__bf16)v[3];
    *reinterpret_cast<v4bf*>(dst + idx) = o;
}

// -------- Kernel 1: per-batch combined 128x128 matrix A[b] --------
// A[b][r,k] = mask[b,r] * ( sum_e ew[b,e]*gains[b,e,k]*chart_e[r,k] - delta_rk )
// chart_e = H(w_e) H(u_e) = I - 2 u u^T - 2 w w^T + 4 (w.u) w u^T  (u,w normalized)
__global__ __launch_bounds__(256) void build_A_kernel(
    const float* __restrict__ logits, const float* __restrict__ mask,
    const float* __restrict__ gains, const float* __restrict__ refl,
    float* __restrict__ Amat) {
    int b = blockIdx.x;
    __shared__ float s_ew[E_];
    __shared__ float s_u[E_][R_];
    __shared__ float s_w[E_][R_];
    __shared__ float s_dot[E_];
    int t = threadIdx.x;

    for (int idx = t; idx < E_ * R_; idx += 256) {
        int e = idx >> 7, r = idx & (R_ - 1);
        s_u[e][r] = refl[(e * 2 + 0) * R_ + r];
        s_w[e][r] = refl[(e * 2 + 1) * R_ + r];
    }
    if (t == 255) {
        float l[E_], m = -1e30f;
#pragma unroll
        for (int e = 0; e < E_; ++e) { l[e] = logits[b * E_ + e] * INV_TEMP; m = fmaxf(m, l[e]); }
        float s = 0.0f;
#pragma unroll
        for (int e = 0; e < E_; ++e) { float ex = expf(l[e] - m); s_ew[e] = ex; s += ex; }
        float inv = 1.0f / s;
#pragma unroll
        for (int e = 0; e < E_; ++e) s_ew[e] *= inv;
    }
    __syncthreads();
    if (t < E_) {
        float nu = 0.0f, nw = 0.0f, d = 0.0f;
        for (int r = 0; r < R_; ++r) {
            float u = s_u[t][r], w = s_w[t][r];
            nu += u * u; nw += w * w; d += u * w;
        }
        float inu = 1.0f / fmaxf(sqrtf(nu), 1e-6f);
        float inw = 1.0f / fmaxf(sqrtf(nw), 1e-6f);
        for (int r = 0; r < R_; ++r) { s_u[t][r] *= inu; s_w[t][r] *= inw; }
        s_dot[t] = d * inu * inw;
    }
    __syncthreads();
    for (int idx = t; idx < R_ * R_; idx += 256) {
        int r = idx >> 7, k = idx & (R_ - 1);
        float diag = (r == k) ? 1.0f : 0.0f;
        float acc = 0.0f;
#pragma unroll
        for (int e = 0; e < E_; ++e) {
            float ur = s_u[e][r], uk = s_u[e][k];
            float wr = s_w[e][r], wk = s_w[e][k];
            float chart = diag - 2.0f * ur * uk - 2.0f * wr * wk + 4.0f * s_dot[e] * wr * uk;
            acc += s_ew[e] * gains[(b * E_ + e) * R_ + k] * chart;
        }
        Amat[(b * R_ + r) * R_ + k] = mask[b * R_ + r] * (acc - diag);
    }
}

// -------- Kernel 2: Wt[b][o][r] = (A[b] @ output_basis)[r,o], bf16, K-contiguous --------
__global__ __launch_bounds__(256) void build_wt_kernel(
    const float* __restrict__ Amat, const float* __restrict__ ob,
    __bf16* __restrict__ wt) {
    int idx = blockIdx.x * 256 + threadIdx.x;
    if (idx >= B_ * O_ * R_) return;
    int b = idx / (O_ * R_);
    int rem = idx - b * (O_ * R_);
    int o = rem >> 7;
    int r = rem & (R_ - 1);
    const float* arow = Amat + (size_t)(b * R_ + r) * R_;
    float acc = 0.0f;
#pragma unroll 4
    for (int k = 0; k < R_; ++k) acc += arow[k] * ob[(size_t)k * O_ + o];
    wt[((size_t)b * O_ + o) * R_ + r] = (__bf16)acc;
}

// -------- Kernel 3: proj[b] = x[b] @ input_basis^T  (M=2048/batch, N=128, K=4096) --------
// 256 blocks of 8 waves; block = 32 rows x 128 cols. Basis k-tile (8KB bf16) fetched with
// CDNA5 async global->LDS copies; x k-tile staged via VGPR cvt -> LDS so HBM reads x once.
// Wave w: row strip rs=w&1 (16 rows), column group cg=w>>2? (w>>1): 32 cols = 2 WMMA tiles.
__global__ __launch_bounds__(256) void proj_kernel(
    const float* __restrict__ x, const __bf16* __restrict__ basisbf,
    __bf16* __restrict__ proj) {
    __shared__ __bf16 s_b[R_ * 32];   // basis tile: 128 rows x 32 k (8 KB)
    __shared__ __bf16 s_x[32 * 32];   // x tile: 32 rows x 32 k (2 KB)
    int b   = blockIdx.x >> 6;        // 64 row-blocks per batch
    int rb  = blockIdx.x & 63;
    int row0 = rb * 32;               // row within batch
    int t = threadIdx.x;
    int wave = t >> 5;
    int rs = wave & 1;                // row strip (16 rows)
    int cg = wave >> 1;               // column group (32 cols)

    // per-thread async-copy source/dest for the basis tile
    int ar  = t >> 1;                 // basis row 0..127
    int akh = (t & 1) * 16;           // k half
    const __bf16* agp = basisbf + (size_t)ar * D_ + akh;
    uint32_t aldsoff = lds_offset_of(s_b + ar * 32 + akh);

    // per-thread x-tile slot: 4 floats
    int xr  = t >> 3;                 // 0..31
    int xkq = (t & 7) * 4;            // 0..28
    const float* xgp = x + ((size_t)b * N_ + row0 + xr) * D_ + xkq;
    v4bf* xls = reinterpret_cast<v4bf*>(s_x + xr * 32 + xkq);

    v8f acc[2] = { v8f_zero(), v8f_zero() };

    for (int k0 = 0; k0 < D_; k0 += 32) {
        // 1) kick off async basis tile copy (tracked by ASYNCcnt)
        async_load_b128x2(aldsoff, agp + k0);
        // 2) stage x tile: streaming (non-temporal) f32 load -> bf16 -> LDS
        v4f xv = __builtin_nontemporal_load(reinterpret_cast<const v4f*>(xgp + k0));
        v4bf xb;
        xb[0] = (__bf16)xv[0]; xb[1] = (__bf16)xv[1];
        xb[2] = (__bf16)xv[2]; xb[3] = (__bf16)xv[3];
        *xls = xb;
        // 3) drain this wave's async copies, then block barrier
        wait_async0();
        __syncthreads();

        v16bf afrag = load_frag_bf16(s_x + rs * 16 * 32, 32);
#pragma unroll
        for (int ct = 0; ct < 2; ++ct) {
            v16bf bfrag = load_frag_bf16(s_b + (cg * 32 + ct * 16) * 32, 32);
            acc[ct] = wmma_bf16(afrag, bfrag, acc[ct]);
        }
        __syncthreads();
    }

    int lane = t & 31;
    int n = lane & 15;
    int mb = (lane >> 4) * 8;
#pragma unroll
    for (int ct = 0; ct < 2; ++ct) {
#pragma unroll
        for (int i = 0; i < 8; ++i) {
            int m = rs * 16 + mb + i;
            proj[((size_t)b * N_ + row0 + m) * R_ + cg * 32 + ct * 16 + n] = (__bf16)acc[ct][i];
        }
    }
}

// -------- Kernel 4: out[b] = proj[b] (2048x128 bf16) @ W[b] (128x4096 bf16) -> f32 --------
// Wave holds full 16x128 A-strip in registers; streams 16 column tiles of 16 (256 cols/wave).
// Output stored non-temporally (write-once, keep L2 for wt/proj).
__global__ __launch_bounds__(256) void out_kernel(
    const __bf16* __restrict__ proj, const __bf16* __restrict__ wt,
    float* __restrict__ out) {
    int b = blockIdx.z;
    int wave = threadIdx.x >> 5;
    int row0 = blockIdx.y * 128 + wave * 16;  // row within batch
    int col0 = blockIdx.x * 256;

    v16bf a[4];
#pragma unroll
    for (int kk = 0; kk < 4; ++kk)
        a[kk] = load_frag_bf16(proj + ((size_t)b * N_ + row0) * R_ + kk * 32, R_);

    int lane = threadIdx.x & 31;
    int n = lane & 15;
    int mb = (lane >> 4) * 8;

    for (int ctile = 0; ctile < 16; ++ctile) {
        int col = col0 + ctile * 16;
        v8f acc = v8f_zero();
#pragma unroll
        for (int kk = 0; kk < 4; ++kk) {
            v16bf bfrag = load_frag_bf16(wt + ((size_t)b * O_ + col) * R_ + kk * 32, R_);
            acc = wmma_bf16(a[kk], bfrag, acc);
        }
#pragma unroll
        for (int i = 0; i < 8; ++i) {
            int m = mb + i;
            __builtin_nontemporal_store(acc[i],
                &out[((size_t)b * N_ + row0 + m) * O_ + col + n]);
        }
    }
}

extern "C" void kernel_launch(void* const* d_in, const int* in_sizes, int n_in,
                              void* d_out, int out_size, void* d_ws, size_t ws_size,
                              hipStream_t stream) {
    (void)in_sizes; (void)n_in; (void)out_size; (void)ws_size;
    const float* x      = (const float*)d_in[0];
    const float* logits = (const float*)d_in[1];
    const float* mask   = (const float*)d_in[2];
    const float* gains  = (const float*)d_in[3];
    const float* ibasis = (const float*)d_in[4];
    const float* obasis = (const float*)d_in[5];
    const float* refl   = (const float*)d_in[6];
    float* out = (float*)d_out;

    char* ws = (char*)d_ws;
    float*  Amat    = (float*)ws;                                    // 256 KB
    __bf16* wt      = (__bf16*)(ws + 262144);                        // 4 MB
    __bf16* proj    = (__bf16*)(ws + 262144 + 4194304);              // 2 MB
    __bf16* basisbf = (__bf16*)(ws + 262144 + 4194304 + 2097152);    // 1 MB

    convert_basis_kernel<<<(R_ * D_) / (256 * 4), 256, 0, stream>>>(ibasis, basisbf);
    build_A_kernel<<<B_, 256, 0, stream>>>(logits, mask, gains, refl, Amat);
    build_wt_kernel<<<(B_ * O_ * R_ + 255) / 256, 256, 0, stream>>>(Amat, obasis, wt);
    proj_kernel<<<B_ * 64, 256, 0, stream>>>(x, basisbf, proj);
    out_kernel<<<dim3(O_ / 256, N_ / 128, B_), 256, 0, stream>>>(proj, wt, out);
}
